// TemporalGNN_57612691309353
// MI455X (gfx1250) — matrix operand order repaired and checked by reading, not compiled, not run on hardware
//
#include <hip/hip_runtime.h>

#define N_NODES 100000
#define N_EDGES 1600000
#define F_IN 64
#define HID 64
#define EF 32
#define C_OUT 4

typedef __attribute__((ext_vector_type(16))) __bf16 v16bf_t;
typedef __attribute__((ext_vector_type(8)))  float  v8f_t;

union BF16Frag {
  v16bf_t v;
  unsigned short u16[16];
  unsigned int   u32[8];
};

// Pack two f32 into one dword of bf16 (truncation) with a single v_perm_b32:
// D = { hi[31:16], lo[31:16] }  (sel bytes: 7,6 from s0=hi ; 3,2 from s1=lo)
__device__ __forceinline__ unsigned int pk_bf16(float lo, float hi) {
  return __builtin_amdgcn_perm(__float_as_uint(hi), __float_as_uint(lo), 0x07060302u);
}

__device__ __forceinline__ unsigned short f2bf_trunc(float f) {
  return (unsigned short)(__float_as_uint(f) >> 16);
}

// Branch-free activations; v_exp_f32 + v_rcp_f32, no IEEE divide expansion.
__device__ __forceinline__ float sigmoid_fast(float x) {
  return __builtin_amdgcn_rcpf(1.0f + __expf(-x));
}
__device__ __forceinline__ float tanh_fast(float x) {
  float t = __expf(-2.0f * fabsf(x));
  float r = (1.0f - t) * __builtin_amdgcn_rcpf(1.0f + t);
  return copysignf(r, x);
}

// A-fragment loader from a bf16 row (16B-aligned, k multiple of 8):
// elements 0..7  -> K = base + 8*half + {0..7}
// elements 8..15 -> K = base + 16 + 8*half + {0..7}
__device__ __forceinline__ void ldA_bf(BF16Frag& A, const unsigned short* base, int half) {
  ((uint4*)A.u32)[0] = *(const uint4*)(base + 8 * half);
  ((uint4*)A.u32)[1] = *(const uint4*)(base + 16 + 8 * half);
}

// Same mapping, gathering f32 and packing to bf16 via v_perm_b32.
__device__ __forceinline__ void ldA_f32(BF16Frag& A, const float* base, int half) {
  float4 a = *(const float4*)(base + 8 * half);
  float4 b = *(const float4*)(base + 8 * half + 4);
  float4 c = *(const float4*)(base + 16 + 8 * half);
  float4 d = *(const float4*)(base + 16 + 8 * half + 4);
  A.u32[0] = pk_bf16(a.x, a.y);
  A.u32[1] = pk_bf16(a.z, a.w);
  A.u32[2] = pk_bf16(b.x, b.y);
  A.u32[3] = pk_bf16(b.z, b.w);
  A.u32[4] = pk_bf16(c.x, c.y);
  A.u32[5] = pk_bf16(c.z, c.w);
  A.u32[6] = pk_bf16(d.x, d.y);
  A.u32[7] = pk_bf16(d.z, d.w);
}

// ---------------- setup kernels ----------------

__global__ void k_init(float* agg, float* deg) {
  int idx = blockIdx.x * blockDim.x + threadIdx.x;
  if (idx < N_NODES * 64) agg[idx] = 0.0f;
  if (idx < N_NODES) deg[idx] = 1.0f;  // self-loop
}

__global__ void k_deg(const int* ei, float* deg) {
  int e = blockIdx.x * blockDim.x + threadIdx.x;
  if (e >= N_EDGES) return;
  unsafeAtomicAdd(&deg[ei[N_EDGES + e]], 1.0f);
}

__global__ void k_dinv(const float* deg, float* dinv) {
  int i = blockIdx.x * blockDim.x + threadIdx.x;
  if (i >= N_NODES) return;
  dinv[i] = rsqrtf(fmaxf(deg[i], 1.0f));
}

// scatter-add: agg[dst] += x[src] * dinv[src]*dinv[dst]; thread = (edge, 8-feature chunk)
__global__ void k_agg(const int* ei, const float* x, const float* dinv, float* agg) {
  int idx = blockIdx.x * blockDim.x + threadIdx.x;
  if (idx >= N_EDGES * 8) return;
  int e = idx >> 3;
  int fb = (idx & 7) * 8;
  int sn = ei[e];
  int dn = ei[N_EDGES + e];
  float nm = dinv[sn] * dinv[dn];
  const float* xr = x + (size_t)sn * 64 + fb;
  float4 a = *(const float4*)(xr);
  float4 b = *(const float4*)(xr + 4);
  float* ar = agg + (size_t)dn * 64 + fb;
  unsafeAtomicAdd(ar + 0, a.x * nm); unsafeAtomicAdd(ar + 1, a.y * nm);
  unsafeAtomicAdd(ar + 2, a.z * nm); unsafeAtomicAdd(ar + 3, a.w * nm);
  unsafeAtomicAdd(ar + 4, b.x * nm); unsafeAtomicAdd(ar + 5, b.y * nm);
  unsafeAtomicAdd(ar + 6, b.z * nm); unsafeAtomicAdd(ar + 7, b.w * nm);
}

// self loop: agg[i] += x[i] * dinv[i]^2 (runs after k_agg, same stream -> ordered)
__global__ void k_self(const float* x, const float* dinv, float* agg) {
  int idx = blockIdx.x * blockDim.x + threadIdx.x;
  if (idx >= N_NODES * 64) return;
  float di = dinv[idx >> 6];
  agg[idx] += x[idx] * di * di;
}

// Fold weights: Mz = Wz @ Wlz[:64], Mh = Wh @ Wlh[:64] (bf16), cz/ch biases, W1 -> bf16
__global__ void k_prep(const float* Wz, const float* bz, const float* Wh, const float* bh,
                       const float* Wlz, const float* blz, const float* Wlh, const float* blh,
                       const float* W1,
                       unsigned short* Mzb, unsigned short* Mhb,
                       float* cz, float* ch, unsigned short* W1bf) {
  int idx = blockIdx.x * blockDim.x + threadIdx.x;
  if (idx < 4096) {
    int k = idx >> 6, j = idx & 63;
    float az = 0.f, ah = 0.f;
    for (int m = 0; m < 64; ++m) {
      az += Wz[k * 64 + m] * Wlz[m * 64 + j];
      ah += Wh[k * 64 + m] * Wlh[m * 64 + j];
    }
    Mzb[idx] = f2bf_trunc(az);
    Mhb[idx] = f2bf_trunc(ah);
  } else if (idx < 4160) {
    int j = idx - 4096;
    float az = blz[j], ah = blh[j];
    for (int m = 0; m < 64; ++m) {
      az += bz[m] * Wlz[m * 64 + j];
      ah += bh[m] * Wlh[m * 64 + j];
    }
    cz[j] = az;
    ch[j] = ah;
  } else if (idx < 4160 + 160 * 64) {
    int i = idx - 4160;
    W1bf[i] = f2bf_trunc(W1[i]);
  }
}

// ---------------- node WMMA kernel ----------------
// Per wave: 16-node x 16-col tile. Z=sigmoid(agg@Mz+cz), Ht=tanh(agg@Mh+ch), h=(1-Z)*Ht -> bf16.
__global__ void k_node(const float* agg, const unsigned short* Mzb, const unsigned short* Mhb,
                       const float* cz, const float* ch, unsigned short* hbf, int ntiles) {
  int tid = threadIdx.x;
  int lane = tid & 31;
  int wave = tid >> 5;       // 0..3, column group
  int n0 = wave * 16;
  int col = lane & 15;
  int half = lane >> 4;

  // B fragments: element i -> K = 32*s + 8*half + (i&7) + 16*(i>>3), N = n0+col
  BF16Frag Bz[2], Bh[2];
#pragma unroll
  for (int s = 0; s < 2; ++s) {
#pragma unroll
    for (int i = 0; i < 16; ++i) {
      int k = 32 * s + 8 * half + (i & 7) + 16 * (i >> 3);
      Bz[s].u16[i] = Mzb[k * 64 + n0 + col];
      Bh[s].u16[i] = Mhb[k * 64 + n0 + col];
    }
  }
  float czv = cz[n0 + col];
  float chv = ch[n0 + col];

  for (int tile = blockIdx.x; tile < ntiles; tile += gridDim.x) {
    int node = tile * 16 + col;              // A-row for this lane
    const float* ar = agg + (size_t)node * 64;
    v8f_t Cz = {0.f, 0.f, 0.f, 0.f, 0.f, 0.f, 0.f, 0.f};
    v8f_t Chm = {0.f, 0.f, 0.f, 0.f, 0.f, 0.f, 0.f, 0.f};
    BF16Frag A;
#pragma unroll
    for (int s = 0; s < 2; ++s) {
      ldA_f32(A, ar + 32 * s, half);
      Cz  = __builtin_amdgcn_wmma_f32_16x16x32_bf16(false, A.v, false, Bz[s].v, (short)0, Cz,  false, false);
      Chm = __builtin_amdgcn_wmma_f32_16x16x32_bf16(false, A.v, false, Bh[s].v, (short)0, Chm, false, false);
    }
#pragma unroll
    for (int r = 0; r < 8; ++r) {
      int node2 = tile * 16 + r + 8 * half;  // D-row
      float z = sigmoid_fast(Cz[r] + czv);
      float t = tanh_fast(Chm[r] + chv);
      hbf[(size_t)node2 * 64 + n0 + col] = f2bf_trunc((1.f - z) * t);
    }
  }
}

// ---------------- edge WMMA kernel ----------------
// emb = [h[src](64) | h[dst](64) | edge_attr(32)]  (K=160)
// hid = relu(emb @ W1 + b1); out = hid @ W2 + b2
__global__ void k_edge(const int* ei, const float* edge_attr,
                       const unsigned short* hbf, const unsigned short* W1bf,
                       const float* b1, const float* W2, const float* b2,
                       float* out, int ntiles) {
  int tid = threadIdx.x;
  int lane = tid & 31;
  int wave = tid >> 5;       // 0..3, column group of hid
  int n0 = wave * 16;
  int col = lane & 15;
  int half = lane >> 4;

  __shared__ float hid_lds[16][65];
  __shared__ float w2_lds[64 * 4];
  __shared__ float b2_lds[4];
  for (int i = tid; i < 256; i += 128) w2_lds[i] = W2[i];
  if (tid < 4) b2_lds[tid] = b2[tid];

  // Preload W1 B-fragments for this wave's 16 columns: 5 K-steps
  BF16Frag Bf[5];
#pragma unroll
  for (int s = 0; s < 5; ++s) {
#pragma unroll
    for (int i = 0; i < 16; ++i) {
      int k = 32 * s + 8 * half + (i & 7) + 16 * (i >> 3);
      Bf[s].u16[i] = W1bf[k * 64 + n0 + col];
    }
  }
  float b1v = b1[n0 + col];
  __syncthreads();

  for (int tile = blockIdx.x; tile < ntiles; tile += gridDim.x) {
    int e = tile * 16 + col;               // A-row -> edge
    int sn = ei[e];
    int dn = ei[N_EDGES + e];
    const unsigned short* hs = hbf + (size_t)sn * 64;
    const unsigned short* hd = hbf + (size_t)dn * 64;
    const float* ea = edge_attr + (size_t)e * 32;

    v8f_t acc = {0.f, 0.f, 0.f, 0.f, 0.f, 0.f, 0.f, 0.f};
    BF16Frag A;
    // K 0..31, 32..63 from h[src]
    ldA_bf(A, hs, half);
    acc = __builtin_amdgcn_wmma_f32_16x16x32_bf16(false, A.v, false, Bf[0].v, (short)0, acc, false, false);
    ldA_bf(A, hs + 32, half);
    acc = __builtin_amdgcn_wmma_f32_16x16x32_bf16(false, A.v, false, Bf[1].v, (short)0, acc, false, false);
    // K 64..127 from h[dst]
    ldA_bf(A, hd, half);
    acc = __builtin_amdgcn_wmma_f32_16x16x32_bf16(false, A.v, false, Bf[2].v, (short)0, acc, false, false);
    ldA_bf(A, hd + 32, half);
    acc = __builtin_amdgcn_wmma_f32_16x16x32_bf16(false, A.v, false, Bf[3].v, (short)0, acc, false, false);
    // K 128..159 from edge_attr (f32 -> bf16 via v_perm)
    ldA_f32(A, ea, half);
    acc = __builtin_amdgcn_wmma_f32_16x16x32_bf16(false, A.v, false, Bf[4].v, (short)0, acc, false, false);

    // bias + relu -> LDS (D layout: row = r + 8*half, col = n0+col)
#pragma unroll
    for (int r = 0; r < 8; ++r) {
      float v = acc[r] + b1v;
      hid_lds[r + 8 * half][n0 + col] = v > 0.f ? v : 0.f;
    }
    __syncthreads();

    // second GEMM: hid(16x64) @ W2(64x4) + b2
    if (tid < 64) {
      int el = tid & 15;
      int c = tid >> 4;
      float s2 = b2_lds[c];
#pragma unroll
      for (int k = 0; k < 64; ++k) s2 += hid_lds[el][k] * w2_lds[k * 4 + c];
      out[(size_t)(tile * 16 + el) * 4 + c] = s2;
    }
    __syncthreads();
  }
}

// ---------------- launcher ----------------

extern "C" void kernel_launch(void* const* d_in, const int* in_sizes, int n_in,
                              void* d_out, int out_size, void* d_ws, size_t ws_size,
                              hipStream_t stream) {
  const float* x         = (const float*)d_in[0];
  const int*   ei        = (const int*)d_in[1];
  const float* edge_attr = (const float*)d_in[2];
  const float* Wz  = (const float*)d_in[3];
  const float* bz  = (const float*)d_in[4];
  // d_in[5..6] (Wr, br) and d_in[11..12] (Wlr, blr): dead — R multiplies H0 == 0
  const float* Wh  = (const float*)d_in[7];
  const float* bh  = (const float*)d_in[8];
  const float* Wlz = (const float*)d_in[9];
  const float* blz = (const float*)d_in[10];
  const float* Wlh = (const float*)d_in[13];
  const float* blh = (const float*)d_in[14];
  const float* W1  = (const float*)d_in[15];
  const float* b1  = (const float*)d_in[16];
  const float* W2  = (const float*)d_in[17];
  const float* b2  = (const float*)d_in[18];
  float* out = (float*)d_out;

  // workspace layout
  char* base = (char*)d_ws;
  size_t off = 0;
  auto take = [&](size_t bytes) {
    char* p = base + off;
    off = (off + bytes + 255) & ~(size_t)255;
    return p;
  };
  float* deg            = (float*)take((size_t)N_NODES * 4);
  float* dinv           = (float*)take((size_t)N_NODES * 4);
  float* agg            = (float*)take((size_t)N_NODES * 64 * 4);
  unsigned short* hbf   = (unsigned short*)take((size_t)N_NODES * 64 * 2);
  unsigned short* Mzb   = (unsigned short*)take(4096 * 2);
  unsigned short* Mhb   = (unsigned short*)take(4096 * 2);
  float* cz             = (float*)take(64 * 4);
  float* ch             = (float*)take(64 * 4);
  unsigned short* W1bf  = (unsigned short*)take(160 * 64 * 2);
  (void)ws_size; (void)in_sizes; (void)n_in; (void)out_size;

  k_init<<<(N_NODES * 64 + 255) / 256, 256, 0, stream>>>(agg, deg);
  k_deg<<<(N_EDGES + 255) / 256, 256, 0, stream>>>(ei, deg);
  k_dinv<<<(N_NODES + 255) / 256, 256, 0, stream>>>(deg, dinv);
  k_agg<<<(N_EDGES * 8 + 255) / 256, 256, 0, stream>>>(ei, x, dinv, agg);
  k_self<<<(N_NODES * 64 + 255) / 256, 256, 0, stream>>>(x, dinv, agg);
  k_prep<<<(4160 + 160 * 64 + 255) / 256, 256, 0, stream>>>(
      Wz, bz, Wh, bh, Wlz, blz, Wlh, blh, W1, Mzb, Mhb, cz, ch, W1bf);
  k_node<<<1024, 128, 0, stream>>>(agg, Mzb, Mhb, cz, ch, hbf, N_NODES / 16);
  k_edge<<<4096, 128, 0, stream>>>(ei, edge_attr, hbf, W1bf, b1, W2, b2, out, N_EDGES / 16);
}
